// WindowSelfAttention_19232863551949
// MI455X (gfx1250) — matrix operand order repaired and checked by reading, not compiled, not run on hardware
//
#include <hip/hip_runtime.h>
#include <hip/hip_bf16.h>

// ---------------------------------------------------------------------------
// CDNA5 (gfx1250) fused window self-attention, f16 WMMA with f32 accumulate.
// TDM (tensor_load_to_lds) stages per-window activations into LDS.
// ---------------------------------------------------------------------------

typedef __attribute__((ext_vector_type(16))) _Float16 v16h;
typedef __attribute__((ext_vector_type(8)))  float    v8f;
typedef unsigned int u32x4 __attribute__((ext_vector_type(4)));
typedef int          i32x4 __attribute__((ext_vector_type(4)));
typedef int          i32x8 __attribute__((ext_vector_type(8)));

#define DIM   512
#define HEADS 8
#define HD    64
#define NTOK  64
#define IMG   192
#define HW    (IMG*IMG)
#define NB    4
#define NWY   24
#define NWX   24
#define NWIN  (NB*NWY*NWX)        // 2304 windows

// workspace layout (bytes)
#define WQKV_OFF   0u
#define WQKV_BYTES (1536u*1024u)          // 96 m-tiles * 16 k-chunks * 1KB
#define WO_OFF     (WQKV_OFF + WQKV_BYTES)
#define WO_BYTES   (512u*1024u)           // 32 m-tiles * 16 k-chunks * 1KB
#define XW_OFF     (WO_OFF + WO_BYTES)    // 2 MB; then 64KB per window

// LDS map (dynamic, 256KB of the WGP's 320KB). Regions are time-multiplexed.
#define LDS_X 0        // X B-fragments (phase1), then P = softmax probs
#define LDS_P 0
#define LDS_Q 65536    // Q canonical [h][tok][d] f16 (phase2), then Oc [tok][c]
#define LDS_O 65536
#define LDS_K 131072   // K canonical [h][tok][d] f16
#define LDS_V 196608   // V canonical [h][d][tok] f16
#define LDS_BYTES 262144

#if defined(__has_builtin)
#if __has_builtin(__builtin_amdgcn_tensor_load_to_lds)
#define HAVE_TDM 1
#endif
#endif

struct F32x8 { uint4 a, b; };

// contiguous 32B fragment (pre-swizzled global frags; B-frags from canonical rows)
static __device__ __forceinline__ v16h ld_frag32(const void* p) {
  F32x8 s;
  s.a = *(const uint4*)p;
  s.b = *(const uint4*)((const char*)p + 16);
  return __builtin_bit_cast(v16h, s);
}
// A-fragment from canonical [row][k] layout: 16B at p (k=hi*16..+7), 16B at p+16
static __device__ __forceinline__ v16h ld_frag_a(const void* p) {
  F32x8 s;
  s.a = *(const uint4*)p;
  s.b = *(const uint4*)((const char*)p + 16);
  return __builtin_bit_cast(v16h, s);
}
static __device__ __forceinline__ unsigned pk2(float x, float y) {
  _Float16 hx = (_Float16)x, hy = (_Float16)y;
  unsigned short ux = __builtin_bit_cast(unsigned short, hx);
  unsigned short uy = __builtin_bit_cast(unsigned short, hy);
  return (unsigned)ux | ((unsigned)uy << 16);
}
static __device__ __forceinline__ unsigned short h16(float x) {
  _Float16 hx = (_Float16)x;
  return __builtin_bit_cast(unsigned short, hx);
}
static __device__ __forceinline__ v8f wmma16(v16h a, v16h b, v8f c) {
  return __builtin_amdgcn_wmma_f32_16x16x32_f16(false, a, false, b, (short)0, c, false, false);
}

// ---------------------------------------------------------------------------
// Kernel 1: convert weights + window-partitioned x into f16 WMMA fragments.
// One wave per 1KB fragment (32 lanes x 16 halves).
//   A-frag element e -> k = hi*16+e                           (weights)
//   B-frag element e -> k = hi*16+e                           (x)
// ---------------------------------------------------------------------------
__global__ void __launch_bounds__(256)
wsa_prep_kernel(const float* __restrict__ x,
                const float* __restrict__ wq,
                const float* __restrict__ wk,
                const float* __restrict__ wv,
                const float* __restrict__ wo,
                char* __restrict__ ws) {
  const int wave = __builtin_amdgcn_readfirstlane(threadIdx.x >> 5);
  const int lane = threadIdx.x & 31;
  const int ra = lane & 15, hi = lane >> 4;
  const int F = blockIdx.x * 8 + wave;
  const int TOTW = 2048;                       // 1536 qkv + 512 wo fragments
  const int TOTAL = TOTW + NWIN * 64;
  if (F >= TOTAL) return;

  unsigned u[8];
  if (F < TOTW) {
    // ---- weight A-fragments ----
    const float* W;
    size_t out_off;
    int r;
    if (F < 1536) {
      int mat = F / 512; r = F % 512;
      W = (mat == 0) ? wq : ((mat == 1) ? wk : wv);
      out_off = WQKV_OFF + (size_t)F * 1024u;
    } else {
      r = F - 1536; W = wo;
      out_off = WO_OFF + (size_t)r * 1024u;
    }
    const int mt = r >> 4, kc = r & 15;
    const int m = mt * 16 + ra;
    const float* p = W + (size_t)m * DIM + kc * 32 + hi * 16;
    float rr[16];
    *(float4*)&rr[0]  = *(const float4*)(p);
    *(float4*)&rr[4]  = *(const float4*)(p + 4);
    *(float4*)&rr[8]  = *(const float4*)(p + 8);
    *(float4*)&rr[12] = *(const float4*)(p + 12);
#pragma unroll
    for (int i = 0; i < 8; i++) u[i] = pk2(rr[2*i], rr[2*i+1]);
    uint4* dst = (uint4*)(ws + out_off + lane * 32);
    dst[0] = *(uint4*)&u[0];
    dst[1] = *(uint4*)&u[4];
  } else {
    // ---- x window B-fragments ----
    const int G = F - TOTW;
    const int w = G >> 6, q = G & 63;
    const int kc = q >> 2, nt = q & 3;
    const int bimg = w / (NWY*NWX);
    const int rem  = w % (NWY*NWX);
    const int wy = rem / NWX, wxw = rem % NWX;
    const int nn = nt * 16 + ra;                   // token in window
    const int hy = wy * 8 + (nn >> 3);
    const int px = wxw * 8 + (nn & 7);
    const int c0 = kc * 32 + hi * 16;
    const float* xp = x + (size_t)bimg * DIM * HW + (size_t)c0 * HW
                        + (size_t)hy * IMG + px;
    float rr[16];
#pragma unroll
    for (int e = 0; e < 16; e++) rr[e] = xp[(size_t)e * HW];
#pragma unroll
    for (int i = 0; i < 8; i++) u[i] = pk2(rr[2*i], rr[2*i+1]);
    uint4* dst = (uint4*)(ws + XW_OFF + (size_t)G * 1024u + lane * 32);
    dst[0] = *(uint4*)&u[0];
    dst[1] = *(uint4*)&u[4];
  }
}

// ---------------------------------------------------------------------------
// Kernel 2: one workgroup (16 waves) per window, fully fused.
// ---------------------------------------------------------------------------
__global__ void __launch_bounds__(512)
wsa_attn_kernel(const char* __restrict__ ws,
                const float* __restrict__ beta,
                const float* __restrict__ bo,
                float* __restrict__ out) {
  extern __shared__ char smem[];
  const int tid  = threadIdx.x;
  const int wave = __builtin_amdgcn_readfirstlane(tid >> 5);
  const int lane = tid & 31;
  const int ra = lane & 15, hi = lane >> 4;
  const int w = blockIdx.x;
  const int bimg = w / (NWY*NWX);
  const int rem  = w % (NWY*NWX);
  const int wy = rem / NWX, wxw = rem % NWX;

  // ---- Phase 0: stage this window's X fragments into LDS (64KB) ----
#if defined(HAVE_TDM)
  if (wave == 0) {
    // Tensor DMA: linear 64KB = one 16384x1 tile of 4-byte elements.
    const unsigned long long ga =
        (unsigned long long)(ws + XW_OFF) + (unsigned long long)w * 65536ull;
    const unsigned lds_base = (unsigned)(size_t)(smem + LDS_X);
    u32x4 g0;
    g0.x = 1u;                                            // count=1, user D#
    g0.y = lds_base;                                      // lds_addr
    g0.z = (unsigned)(ga & 0xffffffffu);                  // global_addr[31:0]
    g0.w = (unsigned)((ga >> 32) & 0x01ffffffu) | (2u << 30); // addr[56:32], type=2
    i32x8 g1 = {};
    g1[0] = (2 << 16);                 // data_size = 4 bytes
    g1[1] = (int)(16384u << 16);       // tensor_dim0[15:0] -> bits 63:48
    g1[2] = (1 << 16);                 // tensor_dim1 = 1
    g1[3] = (int)(16384u << 16);       // tile_dim0 = 16384
    g1[4] = 1;                         // tile_dim1 = 1
    g1[5] = 16384;                     // tensor_dim0_stride
    i32x4 z4 = {};
#if __clang_major__ >= 23
    i32x8 z8 = {};
    __builtin_amdgcn_tensor_load_to_lds(g0, g1, z4, z4, z8, 0);
#else
    __builtin_amdgcn_tensor_load_to_lds(g0, g1, z4, z4, 0);
#endif
    __builtin_amdgcn_s_wait_tensorcnt(0);
  }
#else
  {
    const uint4* src = (const uint4*)(ws + XW_OFF + (size_t)w * 65536u);
    uint4* dst = (uint4*)(smem + LDS_X);
#pragma unroll
    for (int k = 0; k < 8; k++) dst[tid * 8 + k] = src[tid * 8 + k];
  }
#endif
  __syncthreads();

  // ---- Phase 1: QKV = W(1536x512) * X(512x64), 6 m-tiles per wave ----
#pragma unroll 1
  for (int i = 0; i < 6; i++) {
    const int g = wave * 6 + i;                 // 0..95 row-tiles (SGPR)
    const int mat = g >> 5, mt = g & 31;
    v8f acc[4] = {v8f{}, v8f{}, v8f{}, v8f{}};
    const char* wbase = ws + WQKV_OFF + (size_t)g * 16u * 1024u + lane * 32;
#pragma unroll 1
    for (int kc = 0; kc < 16; kc++) {
      __builtin_prefetch(wbase + (kc + 2) * 1024, 0, 1);  // global_prefetch_b8
      v16h a = ld_frag32(wbase + kc * 1024);
      const char* xb = smem + LDS_X + (kc * 4) * 1024 + lane * 32;
      acc[0] = wmma16(a, ld_frag32(xb),          acc[0]);
      acc[1] = wmma16(a, ld_frag32(xb + 1024),   acc[1]);
      acc[2] = wmma16(a, ld_frag32(xb + 2048),   acc[2]);
      acc[3] = wmma16(a, ld_frag32(xb + 3072),   acc[3]);
    }
    const int head = mt >> 2;
    const int d0 = (mt & 3) * 16 + 8 * hi;      // consecutive d for v=0..7
#pragma unroll
    for (int nt = 0; nt < 4; nt++) {
      const int tok = nt * 16 + ra;
      if (mat < 2) {                            // Q,K canonical [h][tok][d]
        unsigned uu[4];
#pragma unroll
        for (int j = 0; j < 4; j++) uu[j] = pk2(acc[nt][2*j], acc[nt][2*j+1]);
        const int base = (mat == 0 ? LDS_Q : LDS_K)
                       + ((head * 64 + tok) * 64 + d0) * 2;
        *(uint4*)(smem + base) = *(uint4*)uu;   // ds_store_b128
      } else {                                  // V canonical [h][d][tok]
#pragma unroll
        for (int j = 0; j < 8; j++) {
          *(unsigned short*)(smem + LDS_V +
              ((head * 64 + (d0 + j)) * 64 + tok) * 2) = h16(acc[nt][j]);
        }
      }
    }
  }
  __syncthreads();

  // ---- Phase 2: attn = softmax(Q^T K * scale + beta), 2 waves per head ----
  {
    const int h = wave >> 1, ntp = wave & 1;
#pragma unroll 1
    for (int t = 0; t < 2; t++) {
      const int nt = ntp * 2 + t;               // query-token tile
      v8f acc[4] = {v8f{}, v8f{}, v8f{}, v8f{}};
#pragma unroll
      for (int kc = 0; kc < 2; kc++) {          // contract over d=64
        v16h a = ld_frag_a(smem + LDS_Q +
            ((h * 64 + nt * 16 + ra) * 64 + kc * 32 + hi * 16) * 2);
#pragma unroll
        for (int m4 = 0; m4 < 4; m4++) {
          v16h bf = ld_frag32(smem + LDS_K +
              ((h * 64 + m4 * 16 + ra) * 64 + kc * 32 + hi * 16) * 2);
          acc[m4] = wmma16(a, bf, acc[m4]);
        }
      }
      // row-wise softmax: row n lives in 16 lanes (same hi half) at vgpr v
#pragma unroll
      for (int v = 0; v < 8; v++) {
        const int n = nt * 16 + v + 8 * hi;
        float val[4];
#pragma unroll
        for (int m4 = 0; m4 < 4; m4++) {
          float bta = beta[(h * 64 + n) * 64 + m4 * 16 + ra];
          val[m4] = acc[m4][v] * 0.125f + bta;
        }
        float mx = fmaxf(fmaxf(val[0], val[1]), fmaxf(val[2], val[3]));
#pragma unroll
        for (int msk = 1; msk < 16; msk <<= 1)
          mx = fmaxf(mx, __shfl_xor(mx, msk, 32));
        float ex[4]; float sum = 0.f;
#pragma unroll
        for (int m4 = 0; m4 < 4; m4++) { ex[m4] = __expf(val[m4] - mx); sum += ex[m4]; }
#pragma unroll
        for (int msk = 1; msk < 16; msk <<= 1)
          sum += __shfl_xor(sum, msk, 32);
        const float inv = 1.0f / sum;
#pragma unroll
        for (int m4 = 0; m4 < 4; m4++) {
          *(unsigned short*)(smem + LDS_P +
              ((h * 64 + n) * 64 + m4 * 16 + ra) * 2) = h16(ex[m4] * inv);
        }
      }
    }
  }
  __syncthreads();

  // ---- Phase 3: O^T[n][d] = P[n][m] * V^T[m][d]; write Oc[tok][c] ----
  {
    const int h = wave >> 1, ntp = wave & 1;
#pragma unroll 1
    for (int t = 0; t < 2; t++) {
      const int nt = ntp * 2 + t;
      v8f acc[4] = {v8f{}, v8f{}, v8f{}, v8f{}};
#pragma unroll
      for (int mc = 0; mc < 2; mc++) {          // contract over m=64
        v16h a = ld_frag_a(smem + LDS_P +
            ((h * 64 + nt * 16 + ra) * 64 + mc * 32 + hi * 16) * 2);
#pragma unroll
        for (int dt = 0; dt < 4; dt++) {
          v16h bf = ld_frag32(smem + LDS_V +
              ((h * 64 + dt * 16 + ra) * 64 + mc * 32 + hi * 16) * 2);
          acc[dt] = wmma16(a, bf, acc[dt]);
        }
      }
#pragma unroll
      for (int dt = 0; dt < 4; dt++) {
        const int c = h * 64 + dt * 16 + ra;    // channel (lane = N = d)
#pragma unroll
        for (int v = 0; v < 8; v++) {
          const int n = nt * 16 + v + 8 * hi;
          *(unsigned short*)(smem + LDS_O + (n * 512 + c) * 2) = h16(acc[dt][v]);
        }
      }
    }
  }
  __syncthreads();

  // ---- Phase 4: out = Wo * O + bo, scatter to [b,c,h,w] ----
#pragma unroll 1
  for (int i = 0; i < 2; i++) {
    const int mt = wave * 2 + i;                // 0..31 output row-tiles
    v8f acc[4] = {v8f{}, v8f{}, v8f{}, v8f{}};
    const char* wbase = ws + WO_OFF + (size_t)(mt * 16) * 1024u + lane * 32;
#pragma unroll 1
    for (int kc = 0; kc < 16; kc++) {
      __builtin_prefetch(wbase + (kc + 2) * 1024, 0, 1);
      v16h a = ld_frag32(wbase + kc * 1024);
#pragma unroll
      for (int nt = 0; nt < 4; nt++) {
        v16h bf = ld_frag32(smem + LDS_O +
            ((nt * 16 + ra) * 512 + kc * 32 + hi * 16) * 2);
        acc[nt] = wmma16(a, bf, acc[nt]);
      }
    }
    const int o0 = mt * 16 + 8 * hi;            // consecutive out channels
    float bb[8];
    *(float4*)&bb[0] = *(const float4*)(bo + o0);
    *(float4*)&bb[4] = *(const float4*)(bo + o0 + 4);
#pragma unroll
    for (int nt = 0; nt < 4; nt++) {
      const int nn = nt * 16 + ra;
      const int hy = wy * 8 + (nn >> 3);
      const int px = wxw * 8 + (nn & 7);
      float* op = out + (size_t)bimg * DIM * HW + (size_t)o0 * HW
                      + (size_t)hy * IMG + px;
#pragma unroll
      for (int v = 0; v < 8; v++) op[(size_t)v * HW] = acc[nt][v] + bb[v];
    }
  }
}

// ---------------------------------------------------------------------------
extern "C" void kernel_launch(void* const* d_in, const int* in_sizes, int n_in,
                              void* d_out, int out_size, void* d_ws, size_t ws_size,
                              hipStream_t stream) {
  const float* x    = (const float*)d_in[0];
  const float* wq   = (const float*)d_in[1];
  const float* wk   = (const float*)d_in[2];
  const float* wv   = (const float*)d_in[3];
  const float* wo   = (const float*)d_in[4];
  const float* bo   = (const float*)d_in[5];
  const float* beta = (const float*)d_in[6];
  char* ws = (char*)d_ws;
  (void)in_sizes; (void)n_in; (void)out_size; (void)ws_size;

  const int total_frags = 2048 + NWIN * 64;     // weight + x fragments
  const int blocks1 = (total_frags + 7) / 8;    // 8 waves (256 thr) per block
  wsa_prep_kernel<<<blocks1, 256, 0, stream>>>(x, wq, wk, wv, wo, ws);
  wsa_attn_kernel<<<NWIN, 512, LDS_BYTES, stream>>>(ws, beta, bo, (float*)d_out);
}
